// LSTM_7653631721534
// MI455X (gfx1250) — matrix operand order repaired and checked by reading, not compiled, not run on hardware
//
#include <hip/hip_runtime.h>
#include <hip/hip_bf16.h>

// ---------------------------------------------------------------------------
// LSTM (B=128, S=512, F=H=512) for gfx1250.
// Phase 1: xb = bf16(x);  P[s][n][b] = x @ Wx_packed + (bi+bh)   (WMMA bf16)
// Phase 2: persistent recurrence, 32 WGs, grid barrier per step  (WMMA bf16)
//          Wh slice preloaded into LDS via Tensor Data Mover (TDM).
// Packed gate order n = g*512 + h, g in {i,f,g,o}.
// ---------------------------------------------------------------------------

typedef __attribute__((ext_vector_type(16))) __bf16 bf16x16;
typedef __attribute__((ext_vector_type(8)))  float  f32x8;
typedef __attribute__((ext_vector_type(4)))  unsigned int u32x4;
typedef __attribute__((ext_vector_type(8)))  int    i32x8;
typedef __attribute__((ext_vector_type(4)))  int    i32x4;

#define B_DIM   128
#define S_DIM   512
#define F_DIM   512
#define H_DIM   512
#define N_PACK  2048          // 4 gates * H
#define NWG_REC 32
#define REC_THREADS 256

#if defined(__has_builtin)
#if __has_builtin(__builtin_amdgcn_tensor_load_to_lds) && \
    __has_builtin(__builtin_amdgcn_s_wait_tensorcnt)
#define HAVE_TDM 1
#endif
#endif
#ifndef HAVE_TDM
#define HAVE_TDM 0
#endif

union FragBF { bf16x16 v; uint4 q[2]; };
union FragF  { f32x8   v; uint4 q[2]; float f[8]; };

__device__ __forceinline__ unsigned short f2bfu(float x) {
  union { float f; unsigned u; } a; a.f = x;
  unsigned r = a.u + 0x7fffu + ((a.u >> 16) & 1u);   // round-to-nearest-even
  return (unsigned short)(r >> 16);
}
__device__ __forceinline__ __bf16 f2bf(float x) {
  union { unsigned short s; __bf16 b; } o; o.s = f2bfu(x);
  return o.b;
}
__device__ __forceinline__ float sigmoidf_(float x) {
  return 1.0f / (1.0f + __expf(-x));
}
__device__ __forceinline__ float tanhf_(float x) {
  float e = __expf(-2.0f * x);
  return (1.0f - e) / (1.0f + e);
}

// ------------------------------ prep kernels -------------------------------

__global__ void prep_misc(const float* __restrict__ bii, const float* __restrict__ bhi,
                          const float* __restrict__ bif, const float* __restrict__ bhf,
                          const float* __restrict__ big, const float* __restrict__ bhg,
                          const float* __restrict__ bio, const float* __restrict__ bho,
                          float* __restrict__ bias, __bf16* __restrict__ hbuf,
                          unsigned* __restrict__ bar) {
  int i = blockIdx.x * blockDim.x + threadIdx.x;
  if (i == 0) *bar = 0u;
  if (i < N_PACK) {
    int g = i >> 9, hc = i & 511;
    const float* bi = (g == 0) ? bii : (g == 1) ? bif : (g == 2) ? big : bio;
    const float* bh = (g == 0) ? bhi : (g == 1) ? bhf : (g == 2) ? bhg : bho;
    bias[i] = bi[hc] + bh[hc];
  }
  for (int j = i; j < 2 * B_DIM * H_DIM; j += gridDim.x * blockDim.x)
    hbuf[j] = f2bf(0.0f);
}

// Pack weights transposed (N-major): WT[n][k] = W_gate[k][hcol], bf16.
__global__ void prep_weights(const float* __restrict__ wii, const float* __restrict__ whi,
                             const float* __restrict__ wif, const float* __restrict__ whf,
                             const float* __restrict__ wig, const float* __restrict__ whg,
                             const float* __restrict__ wio, const float* __restrict__ who,
                             __bf16* __restrict__ wxT, __bf16* __restrict__ whT) {
  int idx = blockIdx.x * blockDim.x + threadIdx.x;
  if (idx >= N_PACK * F_DIM) return;
  int n = idx >> 9, k = idx & 511;
  int g = n >> 9, hc = n & 511;
  const float* wx = (g == 0) ? wii : (g == 1) ? wif : (g == 2) ? wig : wio;
  const float* wh = (g == 0) ? whi : (g == 1) ? whf : (g == 2) ? whg : who;
  wxT[idx] = f2bf(wx[(size_t)k * H_DIM + hc]);
  whT[idx] = f2bf(wh[(size_t)k * H_DIM + hc]);
}

__global__ void prep_x4(const float4* __restrict__ x, ushort4* __restrict__ xb, int n4) {
  int i = blockIdx.x * blockDim.x + threadIdx.x;
  if (i < n4) {
    float4 v = x[i];
    ushort4 o;
    o.x = f2bfu(v.x); o.y = f2bfu(v.y); o.z = f2bfu(v.z); o.w = f2bfu(v.w);
    xb[i] = o;
  }
}

// --------------------- Phase 1: input-side batched GEMM --------------------
// grid = S * 16; block computes a 128(b) x 128(n) tile of P[s] with K=512.
// 8 waves: 4 along M (32 rows), 2 along N (64 cols); wave tile 32x64.

__global__ __launch_bounds__(256) void lstm_xgemm(
    const __bf16* __restrict__ xb,    // [B][S][F] bf16
    const __bf16* __restrict__ wxT,   // [N_PACK][F] bf16 (N-major)
    const float*  __restrict__ bias,  // [N_PACK]
    float* __restrict__ P)            // [S][N_PACK][B] f32
{
  const int bid   = blockIdx.x;
  const int s     = bid >> 4;
  const int ntile = bid & 15;
  const int tid   = threadIdx.x;
  const int wave  = tid >> 5, lane = tid & 31;
  const int lhalf = lane >> 4, l16 = lane & 15;
  const int waveM = wave >> 1, waveN = wave & 1;
  const int b_base = waveM * 32;
  const int n_base = ntile * 128 + waveN * 64;

  f32x8 acc[2][4];
  #pragma unroll
  for (int mf = 0; mf < 2; ++mf)
    #pragma unroll
    for (int nf = 0; nf < 4; ++nf) acc[mf][nf] = (f32x8)(0.0f);

  #pragma unroll 2
  for (int k0 = 0; k0 < F_DIM; k0 += 32) {
    FragBF a[2];
    #pragma unroll
    for (int mf = 0; mf < 2; ++mf) {
      int b = b_base + mf * 16 + l16;
      const __bf16* p = xb + ((size_t)b * S_DIM + s) * F_DIM + k0 + lhalf * 8;
      a[mf].q[0] = *(const uint4*)p;
      a[mf].q[1] = *(const uint4*)(p + 16);
    }
    FragBF bb[4];
    #pragma unroll
    for (int nf = 0; nf < 4; ++nf) {
      int n = n_base + nf * 16 + l16;
      const __bf16* p = wxT + (size_t)n * F_DIM + k0 + lhalf * 8;
      bb[nf].q[0] = *(const uint4*)p;
      bb[nf].q[1] = *(const uint4*)(p + 16);
    }
    #pragma unroll
    for (int mf = 0; mf < 2; ++mf)
      #pragma unroll
      for (int nf = 0; nf < 4; ++nf)
        acc[mf][nf] = __builtin_amdgcn_wmma_f32_16x16x32_bf16(
            false, a[mf].v, false, bb[nf].v, (short)0, acc[mf][nf], false, false);
  }

  // epilogue: +bias, store P[s][n][b] (b contiguous per lane -> 2x b128)
  #pragma unroll
  for (int nf = 0; nf < 4; ++nf) {
    int n = n_base + nf * 16 + l16;
    float bv = bias[n];
    #pragma unroll
    for (int mf = 0; mf < 2; ++mf) {
      FragF o; o.v = acc[mf][nf];
      #pragma unroll
      for (int r = 0; r < 8; ++r) o.f[r] += bv;
      int b0 = b_base + mf * 16 + lhalf * 8;
      float* dst = P + ((size_t)s * N_PACK + n) * B_DIM + b0;
      *(uint4*)dst       = o.q[0];
      *(uint4*)(dst + 4) = o.q[1];
    }
  }
}

// ------------------------- Phase 2: recurrence -----------------------------
// 32 persistent WGs; WG wg owns hidden cols [wg*16, wg*16+16) for all 4 gates.
// Wh slice (64 rows x 512 K, bf16, 64KB) DMA'd into LDS by the TDM and held
// there for all 512 steps. Wave w handles batch rows [w*16, w*16+16); its 4
// accumulators hold i,f,g,o for the same (b,h) per lane, so the cell update
// is register-only; c stays in registers for the whole sequence. h is bf16
// in global (double-buffered, L2-resident); one device barrier per timestep.

__global__ __launch_bounds__(REC_THREADS) void lstm_recur(
    const float*  __restrict__ P,     // [S][N_PACK][B]
    const __bf16* __restrict__ whT,   // [N_PACK][H] bf16 (N-major)
    __bf16* __restrict__ hbuf,        // [2][B][H] bf16
    unsigned* __restrict__ bar,
    float* __restrict__ out)          // h[B][H] then c[B][H]
{
  __shared__ __attribute__((aligned(16))) __bf16 whT_lds[64 * H_DIM]; // 64 KB

  const int wg    = blockIdx.x;     // 0..31
  const int tid   = threadIdx.x;
  const int wave  = tid >> 5, lane = tid & 31;
  const int lhalf = lane >> 4, l16 = lane & 15;
  const int bh    = wave * 16;      // batch-row base for this wave

#if HAVE_TDM
  // One-time TDM preload: 4 tiles (16 rows x 1024 B each), issued by wave 0.
  // D# per CDNA5 ISA 8.3/8.4: group0 = {count=1, lds_addr, global_addr,
  // type=2}; group1 = {data_size=2B, tensor 512x16, tile 512x16, stride 512}.
  if (wave == 0) {
    #pragma unroll
    for (int g = 0; g < 4; ++g) {
      unsigned lds_base = (unsigned)(size_t)(&whT_lds[0]) +
                          (unsigned)(g * 16 * H_DIM * 2);
      unsigned long long ga =
          (unsigned long long)(size_t)(whT + ((size_t)(g * H_DIM + wg * 16)) * H_DIM);
      u32x4 g0;
      g0.x = 1u;                                        // count=1 (valid user D#)
      g0.y = lds_base;                                  // lds_addr (bytes)
      g0.z = (unsigned)(ga & 0xffffffffu);              // global_addr[31:0]
      g0.w = (unsigned)((ga >> 32) & 0x01ffffffu) | (2u << 30); // [56:32] | type=2
      i32x8 g1;
      g1[0] = 0x00010000;          // workgroup_mask=0, data_size=1 (2 bytes)
      g1[1] = (int)(512u << 16);   // tensor_dim0 = 512 elements (lo16)
      g1[2] = (int)(16u << 16);    // tensor_dim0 hi=0, tensor_dim1 = 16 (lo16)
      g1[3] = (int)(512u << 16);   // tensor_dim1 hi=0, tile_dim0 = 512
      g1[4] = 16;                  // tile_dim1 = 16, tile_dim2 = 0
      g1[5] = 512;                 // tensor_dim0_stride[31:0] = 512
      g1[6] = 0;                   // stride hi, tensor_dim1_stride lo
      g1[7] = 0;
      i32x4 z4 = (i32x4)(0);       // 2D tensor: groups 2/3 unused
#if __clang_major__ >= 23
      i32x8 z8 = (i32x8)(0);
      __builtin_amdgcn_tensor_load_to_lds(g0, g1, z4, z4, z8, 0);
#else
      __builtin_amdgcn_tensor_load_to_lds(g0, g1, z4, z4, 0);
#endif
    }
    __builtin_amdgcn_s_wait_tensorcnt(0);
  }
#else
  // Fallback: manual coalesced copy (local row r=g*16+j).
  for (int i = tid; i < 64 * (H_DIM / 8); i += REC_THREADS) {
    int r  = i / (H_DIM / 8);
    int kk = (i % (H_DIM / 8)) * 8;
    int g = r >> 4, j = r & 15;
    const __bf16* src = whT + ((size_t)(g * H_DIM + wg * 16 + j)) * H_DIM + kk;
    *(uint4*)&whT_lds[r * H_DIM + kk] = *(const uint4*)src;
  }
#endif
  __syncthreads();

  f32x8 c_reg = (f32x8)(0.0f);

  for (int s = 0; s < S_DIM; ++s) {
    const __bf16* hcur = hbuf + (size_t)(s & 1) * B_DIM * H_DIM;
    __bf16*       hnxt = hbuf + (size_t)((s + 1) & 1) * B_DIM * H_DIM;

    // init accumulators from precomputed x-side projections (bias included)
    f32x8 acc[4];
    #pragma unroll
    for (int g = 0; g < 4; ++g) {
      int n = g * H_DIM + wg * 16 + l16;
      const float* pp = P + ((size_t)s * N_PACK + n) * B_DIM + bh + lhalf * 8;
      FragF t;
      t.q[0] = *(const uint4*)pp;
      t.q[1] = *(const uint4*)(pp + 4);
      acc[g] = t.v;
    }

    // h @ Wh for all 4 gates, K = 512
    #pragma unroll 4
    for (int k0 = 0; k0 < H_DIM; k0 += 32) {
      FragBF a;
      {
        int b = bh + l16;
        const __bf16* p = hcur + (size_t)b * H_DIM + k0 + lhalf * 8;
        a.q[0] = *(const uint4*)p;
        a.q[1] = *(const uint4*)(p + 16);
      }
      #pragma unroll
      for (int g = 0; g < 4; ++g) {
        FragBF bb;
        const __bf16* p = &whT_lds[(g * 16 + l16) * H_DIM + k0 + lhalf * 8];
        bb.q[0] = *(const uint4*)p;
        bb.q[1] = *(const uint4*)(p + 16);
        acc[g] = __builtin_amdgcn_wmma_f32_16x16x32_bf16(
            false, a.v, false, bb.v, (short)0, acc[g], false, false);
      }
    }

    // cell update: every lane owns i,f,g,o for its 8 (b, hcol) elements
    FragF fi, ff, fg, fo, fc;
    fi.v = acc[0]; ff.v = acc[1]; fg.v = acc[2]; fo.v = acc[3]; fc.v = c_reg;
    const int hcol = wg * 16 + l16;
    #pragma unroll
    for (int r = 0; r < 8; ++r) {
      float i_t = sigmoidf_(fi.f[r]);
      float f_t = sigmoidf_(ff.f[r]);
      float g_t = tanhf_(fg.f[r]);
      float o_t = sigmoidf_(fo.f[r]);
      float c_n = f_t * fc.f[r] + i_t * g_t;
      float h_n = o_t * tanhf_(c_n);
      fc.f[r] = c_n;
      int b = bh + lhalf * 8 + r;
      hnxt[(size_t)b * H_DIM + hcol] = f2bf(h_n);
      if (s == S_DIM - 1) {
        out[(size_t)b * H_DIM + hcol] = h_n;
        out[(size_t)B_DIM * H_DIM + (size_t)b * H_DIM + hcol] = c_n;
      }
    }
    c_reg = fc.v;

    // prefetch next step's gate tile while we wait at the barrier
    if (s + 1 < S_DIM) {
      #pragma unroll
      for (int g = 0; g < 4; ++g) {
        int n = g * H_DIM + wg * 16 + l16;
        __builtin_prefetch(P + ((size_t)(s + 1) * N_PACK + n) * B_DIM + bh + lhalf * 8, 0, 1);
      }
    }

    // device-wide barrier (monotonic counter, reset by prep each launch)
    __threadfence();
    __syncthreads();
    if (tid == 0) {
      atomicAdd(bar, 1u);
      unsigned target = (unsigned)(s + 1) * NWG_REC;
      while (atomicAdd(bar, 0u) < target) __builtin_amdgcn_s_sleep(1);
    }
    __syncthreads();
    __threadfence();
  }
}

// ------------------------------- launcher ----------------------------------

extern "C" void kernel_launch(void* const* d_in, const int* in_sizes, int n_in,
                              void* d_out, int out_size, void* d_ws, size_t ws_size,
                              hipStream_t stream) {
  const float* x   = (const float*)d_in[0];
  const float* wii = (const float*)d_in[1];
  const float* whi = (const float*)d_in[2];
  const float* bii = (const float*)d_in[3];
  const float* bhi = (const float*)d_in[4];
  const float* wif = (const float*)d_in[5];
  const float* whf = (const float*)d_in[6];
  const float* bif = (const float*)d_in[7];
  const float* bhf = (const float*)d_in[8];
  const float* wig = (const float*)d_in[9];
  const float* whg = (const float*)d_in[10];
  const float* big = (const float*)d_in[11];
  const float* bhg = (const float*)d_in[12];
  const float* wio = (const float*)d_in[13];
  const float* who = (const float*)d_in[14];
  const float* bio = (const float*)d_in[15];
  const float* bho = (const float*)d_in[16];

  char* w = (char*)d_ws;
  unsigned* bar  = (unsigned*)(w + 0);                     // 256 B reserved
  float*    bias = (float*)   (w + 256);                   // 8 KB
  __bf16*   hbuf = (__bf16*)  (w + 8448);                  // 256 KB
  __bf16*   wxT  = (__bf16*)  (w + 270592);                // 2 MB
  __bf16*   whT  = (__bf16*)  (w + 2367744);               // 2 MB
  __bf16*   xb   = (__bf16*)  (w + 4464896);               // 64 MB
  float*    P    = (float*)   (w + 71573760);              // 512 MB
  float*    out  = (float*)d_out;

  prep_misc<<<512, 256, 0, stream>>>(bii, bhi, bif, bhf, big, bhg, bio, bho,
                                     bias, hbuf, bar);
  prep_weights<<<(N_PACK * F_DIM + 255) / 256, 256, 0, stream>>>(
      wii, whi, wif, whf, wig, whg, wio, who, wxT, whT);
  const int n4 = (B_DIM * S_DIM * F_DIM) / 4;
  prep_x4<<<(n4 + 255) / 256, 256, 0, stream>>>((const float4*)x, (ushort4*)xb, n4);

  lstm_xgemm<<<S_DIM * 16, 256, 0, stream>>>(xb, wxT, bias, P);
  lstm_recur<<<NWG_REC, REC_THREADS, 0, stream>>>(P, whT, hbuf, bar, out);
}
